// Model_65687229825812
// MI455X (gfx1250) — matrix operand order repaired and checked by reading, not compiled, not run on hardware
//
#include <hip/hip_runtime.h>

#define N_    1024
#define L_    120
#define LQ_   30
#define D_    200
#define P_    5
#define H_    230
#define R_    96
#define EMB_  610
#define EMBP_ 640   // K per conv tap, padded to 20*32
#define LP_   122   // L + 2 zero pad rows
#define HP_   240   // H padded to 15*16
#define KC_   20    // EMBP_/32 k-chunks
#define NT_   15    // HP_/16 n-tiles
#define DC_   7     // ceil(D_/32)

typedef __attribute__((ext_vector_type(16))) __bf16        v16bf;
typedef __attribute__((ext_vector_type(8)))  float         v8f;
typedef __attribute__((ext_vector_type(4)))  unsigned int  u32x4;

__device__ inline unsigned short f2bf(float x) {
    unsigned u = __builtin_bit_cast(unsigned, x);
    u += 0x7FFFu + ((u >> 16) & 1u);      // round-to-nearest-even
    return (unsigned short)(u >> 16);
}

__device__ inline v16bf frag_cast(u32x4 lo, u32x4 hi) {
    struct P { u32x4 a, b; } p{lo, hi};
    return __builtin_bit_cast(v16bf, p);
}

// ---------------------------------------------------------------------------
// Kernel 1: per-sentence BiDAF attention + build padded bf16 im2col input xp
// xp layout: [N][LP_][EMBP_] bf16; row 0 and row 121 are zero (conv padding),
// cols: [0:200)=s  [200:400)=s*s2q  [400:600)=s*q2s  [600:605)=p1 [605:610)=p2
// ---------------------------------------------------------------------------
__global__ __launch_bounds__(256)
void attn_kernel(const int* __restrict__ X, const int* __restrict__ Q,
                 const int* __restrict__ XP1, const int* __restrict__ XP2,
                 const float* __restrict__ word_emb,
                 const float* __restrict__ pos1_emb, const float* __restrict__ pos2_emb,
                 const float* __restrict__ ws_w, const float* __restrict__ ws_b,
                 const float* __restrict__ wq_w, const float* __restrict__ wq_b,
                 const float* __restrict__ wsq_w, const float* __restrict__ wsq_b,
                 unsigned short* __restrict__ xp)
{
    __shared__ float qr[LQ_][DC_ * 32];   // q rows (padded)
    __shared__ float wsv[DC_ * 32];       // ws_w padded
    __shared__ float wsqv[DC_ * 32];      // wsq_w padded
    __shared__ float qdot[LQ_];
    __shared__ int   xidx[L_];
    __shared__ float hm[L_];
    __shared__ float red[8][DC_ * 32];
    __shared__ float q2sv[DC_ * 32];
    __shared__ float bstat[2];

    const int n    = blockIdx.x;
    const int tid  = threadIdx.x;
    const int w    = tid >> 5;
    const int lane = tid & 31;
    const float wsb = ws_b[0], wqb = wq_b[0], wsqb = wsq_b[0];

    for (int i = tid; i < L_; i += 256) xidx[i] = X[n * L_ + i];
    for (int d = tid; d < DC_ * 32; d += 256) {
        wsv[d]  = (d < D_) ? ws_w[d]  : 0.f;
        wsqv[d] = (d < D_) ? wsq_w[d] : 0.f;
    }
    for (int t = tid; t < LQ_ * DC_ * 32; t += 256) {
        int j = t / (DC_ * 32), d = t % (DC_ * 32);
        float v = 0.f;
        if (d < D_) v = word_emb[(size_t)Q[n * LQ_ + j] * D_ + d];
        qr[j][d] = v;
    }
    __syncthreads();
    if (tid < LQ_) {
        float acc = 0.f;
        for (int d = 0; d < D_; ++d) acc += qr[tid][d] * wq_w[d];
        qdot[tid] = acc + wqb;
    }
    __syncthreads();

    // ---- pass 1: per-l attention row, s2q, write s and s*s2q ----
    for (int i = w; i < L_; i += 8) {
        const float* srow = word_emb + (size_t)xidx[i] * D_;
        float s[DC_], tmul[DC_];
        float sd = 0.f;
        #pragma unroll
        for (int c = 0; c < DC_; ++c) {
            int d = c * 32 + lane;
            s[c]    = (d < D_) ? srow[d] : 0.f;
            tmul[c] = s[c] * wsqv[d];
            sd     += s[c] * wsv[d];
        }
        #pragma unroll
        for (int m = 16; m >= 1; m >>= 1) sd += __shfl_xor(sd, m, 32);
        sd += wsb;

        float hj = -3.0e38f;
        for (int j = 0; j < LQ_; ++j) {
            float a = 0.f;
            #pragma unroll
            for (int c = 0; c < DC_; ++c) a += tmul[c] * qr[j][c * 32 + lane];
            #pragma unroll
            for (int m = 16; m >= 1; m >>= 1) a += __shfl_xor(a, m, 32);
            if (lane == j) hj = sd + qdot[j] + a + wsqb;
        }
        float hmx = hj;
        #pragma unroll
        for (int m = 16; m >= 1; m >>= 1) hmx = fmaxf(hmx, __shfl_xor(hmx, m, 32));
        if (lane == 0) hm[i] = hmx;
        float e = (lane < LQ_) ? __expf(hj - hmx) : 0.f;
        float es = e;
        #pragma unroll
        for (int m = 16; m >= 1; m >>= 1) es += __shfl_xor(es, m, 32);
        float a_own = e / es;

        float s2q[DC_];
        #pragma unroll
        for (int c = 0; c < DC_; ++c) s2q[c] = 0.f;
        for (int j = 0; j < LQ_; ++j) {
            float aj = __shfl(a_own, j, 32);
            #pragma unroll
            for (int c = 0; c < DC_; ++c) s2q[c] += aj * qr[j][c * 32 + lane];
        }

        unsigned short* row = xp + (size_t)(n * LP_ + i + 1) * EMBP_;
        #pragma unroll
        for (int c = 0; c < DC_; ++c) {
            int d = c * 32 + lane;
            if (d < D_) {
                row[d]      = f2bf(s[c]);
                row[D_ + d] = f2bf(s[c] * s2q[c]);
            }
        }
        int ip1 = XP1[n * L_ + i], ip2 = XP2[n * L_ + i];
        for (int col = 600 + lane; col < EMBP_; col += 32) {
            float v = 0.f;
            if (col < 605)      v = pos1_emb[ip1 * P_ + (col - 600)];
            else if (col < 610) v = pos2_emb[ip2 * P_ + (col - 605)];
            row[col] = f2bf(v);
        }
    }
    // zero conv-padding rows
    for (int c = tid; c < EMBP_; c += 256) {
        xp[(size_t)(n * LP_) * EMBP_ + c]             = 0;
        xp[(size_t)(n * LP_ + LP_ - 1) * EMBP_ + c]   = 0;
    }
    __syncthreads();

    // ---- pass 2: bvec softmax over l, q2s ----
    if (w == 0) {
        float m = -3.0e38f;
        for (int i = lane; i < L_; i += 32) m = fmaxf(m, hm[i]);
        #pragma unroll
        for (int mm = 16; mm >= 1; mm >>= 1) m = fmaxf(m, __shfl_xor(m, mm, 32));
        if (lane == 0) bstat[0] = m;
    }
    __syncthreads();
    float bmax = bstat[0];
    for (int i = tid; i < L_; i += 256) hm[i] = __expf(hm[i] - bmax);
    __syncthreads();
    if (w == 0) {
        float sgm = 0.f;
        for (int i = lane; i < L_; i += 32) sgm += hm[i];
        #pragma unroll
        for (int mm = 16; mm >= 1; mm >>= 1) sgm += __shfl_xor(sgm, mm, 32);
        if (lane == 0) bstat[1] = sgm;
    }
    __syncthreads();
    float bsum = bstat[1];
    float q2p[DC_];
    #pragma unroll
    for (int c = 0; c < DC_; ++c) q2p[c] = 0.f;
    for (int i = w; i < L_; i += 8) {
        float bv = hm[i] / bsum;
        const float* srow = word_emb + (size_t)xidx[i] * D_;
        #pragma unroll
        for (int c = 0; c < DC_; ++c) {
            int d = c * 32 + lane;
            q2p[c] += bv * ((d < D_) ? srow[d] : 0.f);
        }
    }
    #pragma unroll
    for (int c = 0; c < DC_; ++c) red[w][c * 32 + lane] = q2p[c];
    __syncthreads();
    if (w == 0) {
        #pragma unroll
        for (int c = 0; c < DC_; ++c) {
            float acc = 0.f;
            #pragma unroll
            for (int k = 0; k < 8; ++k) acc += red[k][c * 32 + lane];
            q2sv[c * 32 + lane] = acc;
        }
    }
    __syncthreads();
    // ---- pass 3: write s * q2s ----
    for (int i = w; i < L_; i += 8) {
        const float* srow = word_emb + (size_t)xidx[i] * D_;
        unsigned short* row = xp + (size_t)(n * LP_ + i + 1) * EMBP_;
        #pragma unroll
        for (int c = 0; c < DC_; ++c) {
            int d = c * 32 + lane;
            if (d < D_) row[2 * D_ + d] = f2bf(srow[d] * q2sv[d]);
        }
    }
}

// ---------------------------------------------------------------------------
// Kernel 2: repack conv_w [H][EMB][3] fp32 -> WMMA B fragments bf16
// layout: [(kpos*KC + kc)*NT + nt][lane][8 dwords]; lane<16: K=kc*32+0..15,
// lane>=16: K=kc*32+16..31, col = lane&15, v_i packs (K=kb+2i, kb+2i+1)
// ---------------------------------------------------------------------------
__global__ __launch_bounds__(256)
void packw_kernel(const float* __restrict__ conv_w, unsigned int* __restrict__ wpack)
{
    int idx = blockIdx.x * 256 + threadIdx.x;
    const int total = 3 * KC_ * NT_ * 32 * 8;
    if (idx >= total) return;
    int vi   = idx & 7;
    int lane = (idx >> 3) & 31;
    int nt   = (idx >> 8) % NT_;
    int kk   = (idx >> 8) / NT_;
    int kc   = kk % KC_;
    int kpos = kk / KC_;
    int col  = lane & 15;
    int kb   = (lane < 16) ? 0 : 16;
    int k0   = kc * 32 + kb + 2 * vi;
    int h    = nt * 16 + col;
    float f0 = 0.f, f1 = 0.f;
    if (h < H_) {
        if (k0 < EMB_)     f0 = conv_w[((size_t)h * EMB_ + k0) * 3 + kpos];
        if (k0 + 1 < EMB_) f1 = conv_w[((size_t)h * EMB_ + k0 + 1) * 3 + kpos];
    }
    wpack[idx] = (unsigned)f2bf(f0) | ((unsigned)f2bf(f1) << 16);
}

// ---------------------------------------------------------------------------
// Kernel 3: conv-as-GEMM with WMMA bf16 + fused max-over-time (per l-tile)
// ---------------------------------------------------------------------------
__global__ __launch_bounds__(256)
void conv_gemm_kernel(const unsigned int* __restrict__ xpU,
                      const unsigned int* __restrict__ wpack,
                      float* __restrict__ tmax)
{
    __shared__ unsigned int at[18 * 320];   // 18 rows x 640 bf16 (320 dwords)
    const int n     = blockIdx.y;
    const int lt    = blockIdx.x;
    const int lbase = lt * 16;
    const int tid   = threadIdx.x;
    const int w     = tid >> 5;
    const int lane  = tid & 31;

    for (int idx = tid; idx < 18 * 320; idx += 256) {
        int r = idx / 320, c = idx % 320;
        int xrow = lbase + r;
        unsigned int v = 0;
        if (xrow < LP_) v = xpU[((size_t)n * LP_ + xrow) * 320 + c];
        at[idx] = v;
    }
    __syncthreads();

    const int nt0 = w;
    const int nt1 = w + 8;
    v8f acc0 = {0.f, 0.f, 0.f, 0.f, 0.f, 0.f, 0.f, 0.f};
    v8f acc1 = {0.f, 0.f, 0.f, 0.f, 0.f, 0.f, 0.f, 0.f};
    const int m   = lane & 15;
    const int kbA = (lane < 16) ? 0 : 8;    // dword offset within k-chunk

    for (int kpos = 0; kpos < 3; ++kpos) {
        const int arowbase = (m + kpos) * 320;
        for (int kc = 0; kc < KC_; ++kc) {
            int ab = arowbase + kc * 16 + kbA;
            u32x4 a0 = *(const u32x4*)&at[ab];
            u32x4 a1 = *(const u32x4*)&at[ab + 4];
            v16bf af = frag_cast(a0, a1);

            const unsigned int* bp0 =
                wpack + ((((size_t)(kpos * KC_ + kc)) * NT_ + nt0) * 32 + lane) * 8;
            __builtin_prefetch(bp0 + NT_ * 32 * 8, 0, 1);   // next kc chunk
            u32x4 b0a = *(const u32x4*)&bp0[0];
            u32x4 b0b = *(const u32x4*)&bp0[4];
            v16bf bf0 = frag_cast(b0a, b0b);
            acc0 = __builtin_amdgcn_wmma_f32_16x16x32_bf16(
                false, af, false, bf0, (short)0, acc0, false, false);

            if (nt1 < NT_) {
                const unsigned int* bp1 =
                    wpack + ((((size_t)(kpos * KC_ + kc)) * NT_ + nt1) * 32 + lane) * 8;
                u32x4 b1a = *(const u32x4*)&bp1[0];
                u32x4 b1b = *(const u32x4*)&bp1[4];
                v16bf bf1 = frag_cast(b1a, b1b);
                acc1 = __builtin_amdgcn_wmma_f32_16x16x32_bf16(
                    false, af, false, bf1, (short)0, acc1, false, false);
            }
        }
    }

    // max over valid l rows within this tile (C layout: vgpr r -> M=r(+8))
    const int rowoff = (lane < 16) ? 0 : 8;
    float mv0 = -3.0e38f;
    #pragma unroll
    for (int r = 0; r < 8; ++r) {
        int l = lbase + r + rowoff;
        if (l < L_) mv0 = fmaxf(mv0, acc0[r]);
    }
    mv0 = fmaxf(mv0, __shfl_xor(mv0, 16, 32));
    if (lane < 16) tmax[((size_t)n * 8 + lt) * HP_ + nt0 * 16 + m] = mv0;

    if (nt1 < NT_) {
        float mv1 = -3.0e38f;
        #pragma unroll
        for (int r = 0; r < 8; ++r) {
            int l = lbase + r + rowoff;
            if (l < L_) mv1 = fmaxf(mv1, acc1[r]);
        }
        mv1 = fmaxf(mv1, __shfl_xor(mv1, 16, 32));
        if (lane < 16) tmax[((size_t)n * 8 + lt) * HP_ + nt1 * 16 + m] = mv1;
    }
}

// ---------------------------------------------------------------------------
// Kernel 4: reduce tile maxima + bias + relu -> feat [N][HP_]
// ---------------------------------------------------------------------------
__global__ __launch_bounds__(256)
void feat_kernel(const float* __restrict__ tmax, const float* __restrict__ conv_b,
                 float* __restrict__ feat)
{
    int idx = blockIdx.x * 256 + threadIdx.x;
    if (idx >= N_ * HP_) return;
    int n = idx / HP_, h = idx % HP_;
    float v = -3.0e38f;
    #pragma unroll
    for (int t = 0; t < 8; ++t) v = fmaxf(v, tmax[((size_t)n * 8 + t) * HP_ + h]);
    float out = 0.f;
    if (h < H_) out = fmaxf(v + conv_b[h], 0.f);
    feat[idx] = out;
}

// ---------------------------------------------------------------------------
// Kernel 5: bag self-attention + selective attention + final projection
// ---------------------------------------------------------------------------
__global__ __launch_bounds__(256)
void bag_kernel(const float* __restrict__ feat, const int* __restrict__ X_Scope,
                const int* __restrict__ X_Rel, const float* __restrict__ rel_w,
                const float* __restrict__ rel_b, float* __restrict__ out)
{
    __shared__ float f[8][HP_];
    __shared__ float f2[8][HP_];
    __shared__ float sc[8][8];
    __shared__ float aw[8][8];
    __shared__ float zv[8], wsel[8];
    __shared__ float br[HP_];

    const int b = blockIdx.x;
    const int tid = threadIdx.x;
    const int start = X_Scope[b * 2];

    for (int t = tid; t < 8 * HP_; t += 256) {
        int i = t / HP_, h = t % HP_;
        f[i][h] = feat[(size_t)(start + i) * HP_ + h];
    }
    __syncthreads();
    if (tid < 64) {
        int i = tid >> 3, j = tid & 7;
        float acc = 0.f;
        for (int h = 0; h < H_; ++h) acc += f[i][h] * f[j][h];
        sc[i][j] = acc * 0.0659380473395787f;   // 1/sqrt(230)
    }
    __syncthreads();
    if (tid < 8) {
        float mx = -3.0e38f;
        for (int j = 0; j < 8; ++j) mx = fmaxf(mx, sc[tid][j]);
        float s = 0.f, e[8];
        for (int j = 0; j < 8; ++j) { e[j] = __expf(sc[tid][j] - mx); s += e[j]; }
        for (int j = 0; j < 8; ++j) aw[tid][j] = e[j] / s;
    }
    __syncthreads();
    for (int t = tid; t < 8 * HP_; t += 256) {
        int i = t / HP_, h = t % HP_;
        float acc = 0.f;
        #pragma unroll
        for (int j = 0; j < 8; ++j) acc += aw[i][j] * f[j][h];
        f2[i][h] = acc;
    }
    __syncthreads();
    const int xr = X_Rel[b];
    if (tid < 8) {
        float acc = 0.f;
        const float* rq = rel_w + (size_t)xr * H_;
        for (int h = 0; h < H_; ++h) acc += f2[tid][h] * rq[h];
        zv[tid] = acc;
    }
    __syncthreads();
    if (tid == 0) {
        float mx = -3.0e38f;
        for (int i = 0; i < 8; ++i) mx = fmaxf(mx, zv[i]);
        float s = 0.f, e[8];
        for (int i = 0; i < 8; ++i) { e[i] = __expf(zv[i] - mx); s += e[i]; }
        for (int i = 0; i < 8; ++i) wsel[i] = e[i] / s;
    }
    __syncthreads();
    for (int h = tid; h < HP_; h += 256) {
        float acc = 0.f;
        #pragma unroll
        for (int i = 0; i < 8; ++i) acc += wsel[i] * f2[i][h];
        br[h] = acc;
    }
    __syncthreads();
    if (tid < R_) {
        float acc = 0.f;
        const float* rw = rel_w + (size_t)tid * H_;
        for (int h = 0; h < H_; ++h) acc += br[h] * rw[h];
        out[(size_t)b * R_ + tid] = acc + rel_b[tid];
    }
}

// ---------------------------------------------------------------------------
extern "C" void kernel_launch(void* const* d_in, const int* in_sizes, int n_in,
                              void* d_out, int out_size, void* d_ws, size_t ws_size,
                              hipStream_t stream)
{
    const int*   X        = (const int*)d_in[0];
    const int*   Q        = (const int*)d_in[1];
    const int*   XP1      = (const int*)d_in[2];
    const int*   XP2      = (const int*)d_in[3];
    const int*   X_Scope  = (const int*)d_in[4];
    const int*   X_Rel    = (const int*)d_in[5];
    const float* word_emb = (const float*)d_in[6];
    const float* pos1_emb = (const float*)d_in[7];
    const float* pos2_emb = (const float*)d_in[8];
    const float* ws_w     = (const float*)d_in[9];
    const float* ws_b     = (const float*)d_in[10];
    const float* wq_w     = (const float*)d_in[11];
    const float* wq_b     = (const float*)d_in[12];
    const float* wsq_w    = (const float*)d_in[13];
    const float* wsq_b    = (const float*)d_in[14];
    const float* conv_w   = (const float*)d_in[15];
    const float* conv_b   = (const float*)d_in[16];
    const float* rel_w    = (const float*)d_in[17];
    const float* rel_b    = (const float*)d_in[18];
    float* out = (float*)d_out;

    char* ws = (char*)d_ws;
    size_t off = 0;
    auto carve = [&](size_t bytes) -> void* {
        off = (off + 255) & ~(size_t)255;
        void* p = ws + off;
        off += bytes;
        return p;
    };
    unsigned short* xp    = (unsigned short*)carve((size_t)N_ * LP_ * EMBP_ * 2);
    unsigned int*   wpack = (unsigned int*)carve((size_t)3 * KC_ * NT_ * 32 * 8 * 4);
    float*          tmax  = (float*)carve((size_t)N_ * 8 * HP_ * 4);
    float*          feat  = (float*)carve((size_t)N_ * HP_ * 4);

    packw_kernel<<<dim3((3 * KC_ * NT_ * 32 * 8 + 255) / 256), dim3(256), 0, stream>>>(
        conv_w, wpack);
    attn_kernel<<<dim3(N_), dim3(256), 0, stream>>>(
        X, Q, XP1, XP2, word_emb, pos1_emb, pos2_emb,
        ws_w, ws_b, wq_w, wq_b, wsq_w, wsq_b, xp);
    conv_gemm_kernel<<<dim3(8, N_), dim3(256), 0, stream>>>(
        (const unsigned int*)xp, wpack, tmax);
    feat_kernel<<<dim3((N_ * HP_ + 255) / 256), dim3(256), 0, stream>>>(
        tmax, conv_b, feat);
    bag_kernel<<<dim3(128), dim3(256), 0, stream>>>(
        feat, X_Scope, X_Rel, rel_w, rel_b, out);
}
